// masked_multi_head_attention_88459146428937
// MI455X (gfx1250) — matrix operand order repaired and checked by reading, not compiled, not run on hardware
//
#include <hip/hip_runtime.h>
#include <math.h>

#define S_LEN 2048
#define DMODEL 1024
#define NHEADS 16
#define HD 64
#define MROWS (2 * S_LEN)   // 4096

typedef __attribute__((ext_vector_type(16))) _Float16 v16h;
typedef __attribute__((ext_vector_type(8)))  float    v8f;

union H16 { v16h v; uint4 q[2]; _Float16 h[16]; };

__device__ inline v8f wmma_f16(v16h a, v16h b, v8f c) {
  return __builtin_amdgcn_wmma_f32_16x16x32_f16(false, a, false, b, (short)0, c,
                                                false, false);
}

// A-fragment (16x32 f16): lane holds row = lane&15; halves i -> k = kh+i,
// kh = 16*(lane>>4). Row-major src, stride ld.
__device__ inline v16h load_afrag(const _Float16* base, int ld, int lane) {
  int row = lane & 15;
  int kh  = (lane >> 4) * 16;
  const _Float16* p = base + (size_t)row * ld + kh;
  H16 a;
  a.q[0] = *reinterpret_cast<const uint4*>(p);
  a.q[1] = *reinterpret_cast<const uint4*>(p + 8);
  return a.v;
}

// B-fragment (32x16 f16): lane holds col = lane&15; halves i -> k = kc+i,
// kc = 16*(lane>>4). Source laid out [col][k] (k contiguous), stride ld.
__device__ inline v16h load_bfrag(const _Float16* base, int ld, int lane) {
  int col = lane & 15;
  int kc  = (lane >> 4) * 16;
  const _Float16* p = base + (size_t)col * ld + kc;
  H16 b;
  b.q[0] = *reinterpret_cast<const uint4*>(p);
  b.q[1] = *reinterpret_cast<const uint4*>(p + 8);
  return b.v;
}

// Async-copy one 64(n) x 32(k) f16 W tile (pad stride 40 halves) into LDS.
// 256 16B chunks, 128 threads -> 2 async b128 issues per thread (2/wave).
__device__ inline void issue_wtile_async(const _Float16* __restrict__ Wt, int nb,
                                         int kb, _Float16* buf, int tid) {
  for (int j = 0; j < 2; ++j) {
    int cid = tid + 128 * j;      // 0..255
    int row = cid >> 2;           // 0..63 (n)
    int c   = cid & 3;            // 16B chunk in row
    const _Float16* g = Wt + (size_t)(nb + row) * DMODEL + kb + c * 8;
    unsigned lds = (unsigned)(size_t)(buf + row * 40 + c * 8);
    unsigned long long ga = (unsigned long long)(size_t)g;
    asm volatile("global_load_async_to_lds_b128 %0, %1, off"
                 :: "v"(lds), "v"(ga) : "memory");
  }
}

// One k-step: A-frag from global + 4 B-frags from LDS -> 4 WMMAs.
// B-frags pipelined by one; sched_barrier keeps later loads from hoisting over
// earlier WMMAs.
__device__ inline void gemm_step(const _Float16* Arow, int kb, const _Float16* cur,
                                 int lane, v8f c[4]) {
  v16h a = load_afrag(Arow + kb, DMODEL, lane);
  v16h b = load_bfrag(cur, 40, lane);
#pragma unroll
  for (int t = 0; t < 4; ++t) {
    v16h bn;
    if (t < 3) bn = load_bfrag(cur + (t + 1) * 16 * 40, 40, lane);
    c[t] = wmma_f16(a, b, c[t]);
    __builtin_amdgcn_sched_barrier(0);
    if (t < 3) b = bn;
  }
}

// ---------------------------------------------------------------------------
// Pre-pass A: elementwise fp32 -> f16 (for x, q_in).
// ---------------------------------------------------------------------------
__global__ __launch_bounds__(256)
void cvt_f32_f16_kernel(const float* __restrict__ src, _Float16* __restrict__ dst,
                        int n4) {
  int i = blockIdx.x * 256 + threadIdx.x;
  if (i >= n4) return;
  float4 f = reinterpret_cast<const float4*>(src)[i];
  union { _Float16 h[4]; uint2 u; } o;
  o.h[0] = (_Float16)f.x; o.h[1] = (_Float16)f.y;
  o.h[2] = (_Float16)f.z; o.h[3] = (_Float16)f.w;
  reinterpret_cast<uint2*>(dst)[i] = o.u;
}

// ---------------------------------------------------------------------------
// Pre-pass B: W fp32 [k][n] -> f16 transposed [n][k] (4 matrices via grid.z).
// ---------------------------------------------------------------------------
__global__ __launch_bounds__(256)
void transpose_w_kernel(const float* __restrict__ W0, const float* __restrict__ W1,
                        const float* __restrict__ W2, const float* __restrict__ W3,
                        _Float16* __restrict__ Wt) {
  __shared__ _Float16 t[32][33];
  const int z = blockIdx.z;
  const float* W = (z == 0) ? W0 : (z == 1) ? W1 : (z == 2) ? W2 : W3;
  _Float16* D = Wt + (size_t)z * DMODEL * DMODEL;
  const int kb = blockIdx.y * 32;
  const int nb = blockIdx.x * 32;
  const int tid = threadIdx.x;
  for (int j = 0; j < 4; ++j) {
    int idx = tid + 256 * j;
    int kr = idx >> 5, nc = idx & 31;
    t[nc][kr] = (_Float16)W[(size_t)(kb + kr) * DMODEL + nb + nc];
  }
  __syncthreads();
  for (int j = 0; j < 4; ++j) {
    int idx = tid + 256 * j;
    int r = idx >> 5, nc = idx & 31;
    D[(size_t)(nb + r) * DMODEL + kb + nc] = t[r][nc];
  }
}

// ---------------------------------------------------------------------------
// Kernel 1: Q/K/V projections, async double-buffered weight staging.
// grid = (N/64, M/64, 3); block = 128 (4 waves).
// Q,K -> [B,H,S,HD] f16 ; V -> [B,H,HD,S] f16 (transposed for PV B-frags).
// ---------------------------------------------------------------------------
__global__ __launch_bounds__(128)
void proj_qkv_kernel_impl(const _Float16* __restrict__ Xh,
                          const _Float16* __restrict__ Qinh,
                          const _Float16* __restrict__ Wts,
                          _Float16* __restrict__ Qh, _Float16* __restrict__ Kh,
                          _Float16* __restrict__ Vt) {
  __shared__ __align__(16) _Float16 Wbuf0[64 * 40];
  __shared__ __align__(16) _Float16 Wbuf1[64 * 40];
  const int mode = blockIdx.z;                    // 0=Q,1=K,2=V
  const _Float16* A  = (mode == 0) ? Qinh : Xh;
  const _Float16* Wt = Wts + (size_t)mode * DMODEL * DMODEL;
  const int nb = blockIdx.x * 64;
  const int mb = blockIdx.y * 64;
  const int tid = threadIdx.x;
  const int lane = tid & 31;
  const int w = tid >> 5;
  const int half = lane >> 4;
  const int nlo = lane & 15;

  v8f c[4] = {};
  const _Float16* Arow = A + (size_t)(mb + w * 16) * DMODEL;

  constexpr int NI = DMODEL / 32;
  issue_wtile_async(Wt, nb, 0, Wbuf0, tid);
  _Float16* cur = Wbuf0;
  _Float16* nxt = Wbuf1;
#pragma clang loop unroll(disable)
  for (int i = 0; i < NI - 1; ++i) {
    issue_wtile_async(Wt, nb, (i + 1) * 32, nxt, tid);
    asm volatile("s_wait_asynccnt 0x2" ::: "memory");  // tile i landed
    __syncthreads();
    gemm_step(Arow, i * 32, cur, lane, c);
    _Float16* tmp = cur; cur = nxt; nxt = tmp;
  }
  asm volatile("s_wait_asynccnt 0x0" ::: "memory");
  __syncthreads();
  gemm_step(Arow, (NI - 1) * 32, cur, lane, c);

  if (mode == 2) {
    for (int t = 0; t < 4; ++t) {
      int col = nb + t * 16 + nlo;
      int h = col >> 6, d = col & 63;
      for (int r = 0; r < 8; ++r) {
        int row = mb + w * 16 + r + 8 * half;
        int bb = row >> 11, s = row & (S_LEN - 1);
        Vt[(((size_t)bb * NHEADS + h) * HD + d) * S_LEN + s] = (_Float16)c[t][r];
      }
    }
  } else {
    _Float16* dst = (mode == 0) ? Qh : Kh;
    for (int t = 0; t < 4; ++t) {
      int col = nb + t * 16 + nlo;
      int h = col >> 6, d = col & 63;
      for (int r = 0; r < 8; ++r) {
        int row = mb + w * 16 + r + 8 * half;
        int bb = row >> 11, s = row & (S_LEN - 1);
        dst[(((size_t)bb * NHEADS + h) * S_LEN + s) * HD + d] = (_Float16)c[t][r];
      }
    }
  }
}

// ---------------------------------------------------------------------------
// Kernel 2: causal flash attention. grid = (B*H, S/64); block = 128 (4 waves).
// softmax((QK^T + mask)/32) matched via mask-to--inf then *1/32.
// ---------------------------------------------------------------------------
__global__ __launch_bounds__(128)
void flash_attn_kernel(const _Float16* __restrict__ Qh, const _Float16* __restrict__ Kh,
                       const _Float16* __restrict__ Vt, _Float16* __restrict__ ctx) {
  __shared__ __align__(16) _Float16 Pt[4][16 * 40];
  const int bh = blockIdx.x;
  const int qb = blockIdx.y * 64;
  const int tid = threadIdx.x;
  const int lane = tid & 31;
  const int w = tid >> 5;
  const int half = lane >> 4;
  const int nlo = lane & 15;
  const int qw = qb + w * 16;
  const int hh = bh & (NHEADS - 1);
  const int bb = bh >> 4;

  const _Float16* Q = Qh + (size_t)bh * S_LEN * HD;
  const _Float16* K = Kh + (size_t)bh * S_LEN * HD;
  const _Float16* V = Vt + (size_t)bh * HD * S_LEN;

  v16h aq0 = load_afrag(Q + (size_t)qw * HD, HD, lane);
  v16h aq1 = load_afrag(Q + (size_t)qw * HD + 32, HD, lane);

  v8f O[4] = {};
  float m[8], l[8];
  const float NEG_INF = -__builtin_inff();
  for (int r = 0; r < 8; ++r) { m[r] = NEG_INF; l[r] = 0.f; }
  const float inv = 0.03125f;
  _Float16* P = Pt[w];

#pragma clang loop unroll(disable)
  for (int jb = 0; jb < qw + 16; jb += 32) {
    v8f s0 = {}, s1 = {};
    {
      v16h b0 = load_bfrag(K + (size_t)jb * HD, HD, lane);
      v16h b1 = load_bfrag(K + (size_t)jb * HD + 32, HD, lane);
      s0 = wmma_f16(aq0, b0, s0);
      __builtin_amdgcn_sched_barrier(0);
      v16h b2 = load_bfrag(K + (size_t)(jb + 16) * HD, HD, lane);
      s0 = wmma_f16(aq1, b1, s0);
      __builtin_amdgcn_sched_barrier(0);
      v16h b3 = load_bfrag(K + (size_t)(jb + 16) * HD + 32, HD, lane);
      s1 = wmma_f16(aq0, b2, s1);
      __builtin_amdgcn_sched_barrier(0);
      s1 = wmma_f16(aq1, b3, s1);
      __builtin_amdgcn_sched_barrier(0);
    }

    int k0 = jb + nlo, k1 = k0 + 16;
    float mt[8];
    for (int r = 0; r < 8; ++r) {
      int q = qw + r + 8 * half;
      float v0 = (k0 > q) ? NEG_INF : s0[r] * inv;
      float v1 = (k1 > q) ? NEG_INF : s1[r] * inv;
      s0[r] = v0; s1[r] = v1;
      mt[r] = fmaxf(v0, v1);
    }
    for (int msk = 1; msk <= 8; msk <<= 1)
      for (int r = 0; r < 8; ++r)
        mt[r] = fmaxf(mt[r], __shfl_xor(mt[r], msk, 32));

    float ls[8];
    for (int r = 0; r < 8; ++r) {
      float mn = fmaxf(m[r], mt[r]);
      float alpha = __expf(m[r] - mn);
      m[r] = mn;
      float p0 = __expf(s0[r] - mn);
      float p1 = __expf(s1[r] - mn);
      s0[r] = p0; s1[r] = p1;
      ls[r] = p0 + p1;
      l[r] *= alpha;
      for (int t = 0; t < 4; ++t) O[t][r] *= alpha;
    }
    for (int msk = 1; msk <= 8; msk <<= 1)
      for (int r = 0; r < 8; ++r)
        ls[r] += __shfl_xor(ls[r], msk, 32);
    for (int r = 0; r < 8; ++r) l[r] += ls[r];

    for (int r = 0; r < 8; ++r) {
      int row = r + 8 * half;
      P[row * 40 + nlo]      = (_Float16)s0[r];
      P[row * 40 + 16 + nlo] = (_Float16)s1[r];
    }
    asm volatile("s_wait_dscnt 0x0" ::: "memory");
    __builtin_amdgcn_wave_barrier();
    v16h ap = load_afrag(P, 40, lane);
    {
      v16h bv = load_bfrag(V + jb, S_LEN, lane);
#pragma unroll
      for (int t = 0; t < 4; ++t) {
        v16h bn;
        if (t < 3) bn = load_bfrag(V + (size_t)(t + 1) * 16 * S_LEN + jb, S_LEN, lane);
        O[t] = wmma_f16(ap, bv, O[t]);
        __builtin_amdgcn_sched_barrier(0);
        if (t < 3) bv = bn;
      }
    }
    __builtin_amdgcn_wave_barrier();
  }

  for (int t = 0; t < 4; ++t) {
    int col = hh * HD + t * 16 + nlo;
    for (int r = 0; r < 8; ++r) {
      int s = qb + w * 16 + r + 8 * half;
      float o = O[t][r] / l[r];
      ctx[((size_t)bb * S_LEN + s) * DMODEL + col] = (_Float16)o;
    }
  }
}

// ---------------------------------------------------------------------------
// Kernel 3: out = ctx(f16) @ WO -> fp32, async double-buffered weight staging.
// ---------------------------------------------------------------------------
__global__ __launch_bounds__(128)
void out_proj_kernel(const _Float16* __restrict__ ctx, const _Float16* __restrict__ WOt,
                     float* __restrict__ out) {
  __shared__ __align__(16) _Float16 Wbuf0[64 * 40];
  __shared__ __align__(16) _Float16 Wbuf1[64 * 40];
  const int nb = blockIdx.x * 64;
  const int mb = blockIdx.y * 64;
  const int tid = threadIdx.x;
  const int lane = tid & 31;
  const int w = tid >> 5;
  const int half = lane >> 4;
  const int nlo = lane & 15;

  v8f c[4] = {};
  const _Float16* Arow = ctx + (size_t)(mb + w * 16) * DMODEL;

  constexpr int NI = DMODEL / 32;
  issue_wtile_async(WOt, nb, 0, Wbuf0, tid);
  _Float16* cur = Wbuf0;
  _Float16* nxt = Wbuf1;
#pragma clang loop unroll(disable)
  for (int i = 0; i < NI - 1; ++i) {
    issue_wtile_async(WOt, nb, (i + 1) * 32, nxt, tid);
    asm volatile("s_wait_asynccnt 0x2" ::: "memory");
    __syncthreads();
    gemm_step(Arow, i * 32, cur, lane, c);
    _Float16* tmp = cur; cur = nxt; nxt = tmp;
  }
  asm volatile("s_wait_asynccnt 0x0" ::: "memory");
  __syncthreads();
  gemm_step(Arow, (NI - 1) * 32, cur, lane, c);

  for (int t = 0; t < 4; ++t) {
    int col = nb + t * 16 + nlo;
    for (int r = 0; r < 8; ++r) {
      int row = mb + w * 16 + r + 8 * half;
      out[(size_t)row * DMODEL + col] = c[t][r];
    }
  }
}

extern "C" void kernel_launch(void* const* d_in, const int* in_sizes, int n_in,
                              void* d_out, int out_size, void* d_ws, size_t ws_size,
                              hipStream_t stream) {
  (void)in_sizes; (void)n_in; (void)out_size; (void)ws_size;
  const float* x    = (const float*)d_in[0];
  const float* q_in = (const float*)d_in[1];
  const float* WQ   = (const float*)d_in[2];
  const float* WK   = (const float*)d_in[3];
  const float* WV   = (const float*)d_in[4];
  const float* WO   = (const float*)d_in[5];
  float* out = (float*)d_out;

  const size_t perBH = (size_t)2 * NHEADS * S_LEN * HD;  // 4M f16 elems
  const size_t perAct = (size_t)MROWS * DMODEL;          // 4M f16 elems
  _Float16* Qh   = (_Float16*)d_ws;
  _Float16* Kh   = Qh + perBH;
  _Float16* Vt   = Kh + perBH;
  _Float16* ctx  = Vt + perBH;
  _Float16* Xh   = ctx + perBH;
  _Float16* Qinh = Xh + perAct;
  _Float16* Wts  = Qinh + perAct;   // 4 x [1024][1024] f16: WQt,WKt,WVt,WOt

  const int n4 = (int)(perAct / 4);
  cvt_f32_f16_kernel<<<(n4 + 255) / 256, 256, 0, stream>>>(x, Xh, n4);
  cvt_f32_f16_kernel<<<(n4 + 255) / 256, 256, 0, stream>>>(q_in, Qinh, n4);

  dim3 gT(DMODEL / 32, DMODEL / 32, 4);
  transpose_w_kernel<<<gT, 256, 0, stream>>>(WQ, WK, WV, WO, Wts);

  dim3 gProj(DMODEL / 64, MROWS / 64, 3);
  proj_qkv_kernel_impl<<<gProj, 128, 0, stream>>>(Xh, Qinh, Wts, Qh, Kh, Vt);

  dim3 gAttn(2 * NHEADS, S_LEN / 64);
  flash_attn_kernel<<<gAttn, 128, 0, stream>>>(Qh, Kh, Vt, ctx);

  dim3 gOut(DMODEL / 64, MROWS / 64);
  out_proj_kernel<<<gOut, 128, 0, stream>>>(ctx, Wts + (size_t)3 * DMODEL * DMODEL, out);
}
